// MultiHeadAttention_25237227831356
// MI455X (gfx1250) — matrix operand order repaired and checked by reading, not compiled, not run on hardware
//
#include <hip/hip_runtime.h>

typedef __attribute__((ext_vector_type(16))) _Float16 v16h;
typedef __attribute__((ext_vector_type(8)))  float    v8f;

#define NB   4
#define SEQ  2048
#define NH   8
#define HD   64
#define ED   512

// ---------------- WMMA helpers (CDNA5: wave32, 16x16x32 f16 -> f32) ----------------
__device__ __forceinline__ v8f wmma16(v16h a, v16h b, v8f c) {
  // (neg_a, A, neg_b, B, c_mod, C, reuse_a, reuse_b)
  return __builtin_amdgcn_wmma_f32_16x16x32_f16(false, a, false, b, (short)0, c, false, false);
}

// A fragment 16x32 f16: lane l holds row m = l&15.
//   elems i -> k = kb + ((l>>4)*16) + i
__device__ __forceinline__ v16h loadA_f32(const float* __restrict__ src, int m0, int ld,
                                          int kb, int lane) {
  const int m  = lane & 15;
  const int ko = (lane >> 4) << 4;
  const float* p = src + (size_t)(m0 + m) * ld + kb + ko;
  v16h f;
#pragma unroll
  for (int i = 0; i < 8; ++i) {
    f[i]     = (_Float16)p[i];
    f[i + 8] = (_Float16)p[i + 8];
  }
  return f;
}
__device__ __forceinline__ v16h loadA_f16(const _Float16* __restrict__ src, int m0, int ld,
                                          int kb, int lane) {
  const int m  = lane & 15;
  const int ko = (lane >> 4) << 4;
  const _Float16* p = src + (size_t)(m0 + m) * ld + kb + ko;
  v16h f;
#pragma unroll
  for (int i = 0; i < 8; ++i) {
    f[i]     = p[i];
    f[i + 8] = p[i + 8];
  }
  return f;
}

// B fragment 32x16: lane l holds column n = l&15; elems i -> k = kb + ((l>>4)*16) + i.
// "Transposed" source: src row-major [n][k] (weights W[out][in]; B = W^T).
__device__ __forceinline__ v16h loadBT_f32(const float* __restrict__ src, int n0, int ld,
                                           int kb, int lane) {
  const int n  = lane & 15;
  const int kh = (lane >> 4) << 4;
  const float* p = src + (size_t)(n0 + n) * ld + kb + kh;
  v16h f;
#pragma unroll
  for (int i = 0; i < 16; ++i) f[i] = (_Float16)p[i];
  return f;
}
__device__ __forceinline__ v16h loadBT_f16(const _Float16* __restrict__ src, int n0, int ld,
                                           int kb, int lane) {
  const int n  = lane & 15;
  const int kh = (lane >> 4) << 4;
  const _Float16* p = src + (size_t)(n0 + n) * ld + kb + kh;
  v16h f;
#pragma unroll
  for (int i = 0; i < 16; ++i) f[i] = p[i];
  return f;
}
// Row-major B source: src is [k][n] (projected V tiles; works for global or LDS pointers).
__device__ __forceinline__ v16h loadB_rm_f16(const _Float16* __restrict__ src, int kb, int ld,
                                             int n0, int lane) {
  const int n  = lane & 15;
  const int kh = (lane >> 4) << 4;
  const _Float16* p = src + (size_t)(kb + kh) * ld + n0 + n;
  v16h f;
#pragma unroll
  for (int i = 0; i < 16; ++i) f[i] = p[(size_t)i * ld];
  return f;
}

// C/D 16x16 f32: lane l (<16): VGPR j -> [m0+j][n0+l]; lane l (>=16): [m0+8+j][n0+l-16]
__device__ __forceinline__ void storeC_f16(_Float16* __restrict__ dst, v8f c, int m0, int n0,
                                           int ld, int lane) {
  const int col = n0 + (lane & 15);
  const int r0  = m0 + ((lane >> 4) << 3);
#pragma unroll
  for (int j = 0; j < 8; ++j) dst[(size_t)(r0 + j) * ld + col] = (_Float16)c[j];
}

// ---------------- 1) per-head projection: y = x_head @ W^T  (f32 in, f16 out) ------
// grid: (B*S/16, NH), block: 32 (one wave). Output layout [b,h,s,64].
__global__ void __launch_bounds__(32)
proj_kernel(const float* __restrict__ x, const float* __restrict__ W,
            _Float16* __restrict__ dst) {
  const int lane = threadIdx.x;
  const int row0 = blockIdx.x << 4;   // over B*S
  const int h    = blockIdx.y;
  v8f acc[4] = {};
#pragma unroll
  for (int kb = 0; kb < HD; kb += 32) {
    v16h a = loadA_f32(x + h * HD, row0, ED, kb, lane);
#pragma unroll
    for (int nt = 0; nt < 4; ++nt) {
      v16h b = loadBT_f32(W, nt * 16, HD, kb, lane);
      acc[nt] = wmma16(a, b, acc[nt]);
    }
  }
  const int bi = row0 / SEQ;
  const int s0 = row0 - bi * SEQ;
  _Float16* out = dst + ((size_t)(bi * NH + h) * SEQ + s0) * HD;
#pragma unroll
  for (int nt = 0; nt < 4; ++nt) storeC_f16(out, acc[nt], 0, nt * 16, HD, lane);
}

// ---------------- 2) fused energy + mask + softmax -----------------------------------
// One block = one (b,h) and 16 query rows x full 2048 keys, strip lives in LDS (128KB).
// 8 waves: wave w computes k-tiles t = w, w+8, ..., w+120 (keys 16t..16t+15). Attention written to HBM once.
// grid: (S/16, B*NH), block: 256, dynamic LDS = 16*SEQ*4 bytes.
__global__ void __launch_bounds__(256)
scores_softmax_kernel(const _Float16* __restrict__ qh, const _Float16* __restrict__ kh,
                      const int* __restrict__ mask, float* __restrict__ att) {
  extern __shared__ float strip[];        // [16][SEQ]
  __shared__ float part[16][16];
  __shared__ float rowred[16];
  const int tid  = threadIdx.x;
  const int lane = tid & 31;
  const int wv   = tid >> 5;              // 0..7
  const int q0   = blockIdx.x << 4;
  const int bh   = blockIdx.y;
  const int b    = bh >> 3;               // NH == 8
  const _Float16* qb = qh + (size_t)bh * SEQ * HD;
  const _Float16* kb = kh + (size_t)bh * SEQ * HD;

  // Q fragments loaded once per wave, reused across all 16 of its k-tiles
  const v16h a0 = loadA_f16(qb, q0, HD, 0, lane);
  const v16h a1 = loadA_f16(qb, q0, HD, 32, lane);

  const float scale = 0.044194173824159216f;  // 1/sqrt(512)
  const int colL = lane & 15;
  const int r0   = (lane >> 4) << 3;
  for (int t = wv; t < SEQ / 16; t += 8) {
    const int k0 = t << 4;
    v8f acc = {};
    acc = wmma16(a0, loadBT_f16(kb, k0, HD, 0, lane), acc);   // B = K^T
    acc = wmma16(a1, loadBT_f16(kb, k0, HD, 32, lane), acc);
    const int col = k0 + colL;
    const int mv  = mask[b * SEQ + col];
#pragma unroll
    for (int j = 0; j < 8; ++j) {
      float e = (mv == 0) ? -1.0e20f : acc[j];  // mask before scale (reference order)
      strip[(r0 + j) * SEQ + col] = e * scale;
    }
  }
  __syncthreads();

  // per-row softmax over the strip: thread t owns row t>>4, 128-col segment t&15
  const int row = tid >> 4;
  const int seg = tid & 15;
  float* rp = strip + row * SEQ + seg * 128;
  float mx = rp[0];
  for (int i = 1; i < 128; ++i) mx = fmaxf(mx, rp[i]);
  part[row][seg] = mx;
  __syncthreads();
  if (tid < 16) {
    float m = part[tid][0];
#pragma unroll
    for (int i = 1; i < 16; ++i) m = fmaxf(m, part[tid][i]);
    rowred[tid] = m;
  }
  __syncthreads();
  mx = rowred[row];
  float sum = 0.f;
  for (int i = 0; i < 128; ++i) {
    float e = __expf(rp[i] - mx);
    rp[i] = e;
    sum += e;
  }
  part[row][seg] = sum;
  __syncthreads();
  if (tid < 16) {
    float s = part[tid][0];
#pragma unroll
    for (int i = 1; i < 16; ++i) s += part[tid][i];
    rowred[tid] = 1.0f / s;
  }
  __syncthreads();
  const float inv = rowred[row];
  float* op = att + (size_t)bh * SEQ * SEQ + (size_t)(q0 + row) * SEQ + seg * 128;
  for (int i = 0; i < 128; i += 4) {
    float4 v4 = make_float4(rp[i] * inv, rp[i + 1] * inv, rp[i + 2] * inv, rp[i + 3] * inv);
    *(float4*)(op + i) = v4;
  }
}

// ---------------- 3) head_out = attention @ V ---------------------------------------
// 4 waves per block share V chunks double-buffered through LDS via async-to-LDS copies
// (ASYNCcnt path). grid: (S/64, B*NH), block: 128. Output f16 [b*s][512] (head concat).
__global__ void __launch_bounds__(128)
av_kernel(const float* __restrict__ att, const _Float16* __restrict__ vh,
          _Float16* __restrict__ oh) {
  __shared__ _Float16 vbuf[2][32][HD];      // 2 x 4KB chunks of V
  const int tid  = threadIdx.x;
  const int lane = tid & 31;
  const int wv   = tid >> 5;                // 0..3
  const int q0   = (blockIdx.x << 6) + (wv << 4);
  const int bh   = blockIdx.y;
  const int b = bh >> 3, h = bh & 7;
  const float* ab = att + (size_t)bh * SEQ * SEQ;
  const _Float16* vb = vh + (size_t)bh * SEQ * HD;

  // async copy mapping: 128 threads x 2 issues x 16B cover the 32x64 f16 chunk
  const int cr = tid >> 3;                  // copy row 0..15 (and +16)
  const int ce = (tid & 7) << 3;            // element col 0..56 step 8 (16B)
  auto issue = [&](int chunk, int bf) {
    const _Float16* g0 = vb + (size_t)(chunk * 32 + cr) * HD + ce;
    const _Float16* g1 = g0 + (size_t)16 * HD;
    unsigned l0 = (unsigned)(uintptr_t)(void*)&vbuf[bf][cr][ce];
    unsigned l1 = (unsigned)(uintptr_t)(void*)&vbuf[bf][cr + 16][ce];
    asm volatile("global_load_async_to_lds_b128 %0, %1, off"
                 :: "v"(l0), "v"(g0) : "memory");
    asm volatile("global_load_async_to_lds_b128 %0, %1, off"
                 :: "v"(l1), "v"(g1) : "memory");
  };

  v8f acc[4] = {};
  issue(0, 0);
  const int nChunks = SEQ / 32;
  for (int i = 0; i < nChunks; ++i) {
    asm volatile("s_wait_asynccnt 0x0" ::: "memory");  // own copies for chunk i landed
    __syncthreads();                                   // chunk i complete; buf (i+1)&1 free
    if (i + 1 < nChunks) issue(i + 1, (i + 1) & 1);
    const int kk = i << 5;
    if (kk + 32 < SEQ)  // stream-hint next attention strip (global_prefetch_b8)
      __builtin_prefetch(&ab[(size_t)(q0 + (lane & 15)) * SEQ + kk + 32], 0, 0);
    v16h a = loadA_f32(ab, q0, SEQ, kk, lane);
    const _Float16* vt = &vbuf[i & 1][0][0];
#pragma unroll
    for (int nt = 0; nt < 4; ++nt) {
      v16h bt = loadB_rm_f16(vt, 0, HD, nt * 16, lane);  // ds reads feeding WMMA
      acc[nt] = wmma16(a, bt, acc[nt]);
    }
  }
  _Float16* out = oh + (size_t)(b * SEQ + q0) * ED + h * HD;
#pragma unroll
  for (int nt = 0; nt < 4; ++nt) storeC_f16(out, acc[nt], 0, nt * 16, ED, lane);
}

// ---------------- 4) out = head_out @ Wo^T + bo -------------------------------------
// grid: (B*S/16, ED/64), block: 32; K loop over 512.
__global__ void __launch_bounds__(32)
outproj_kernel(const _Float16* __restrict__ oh, const float* __restrict__ Wo,
               const float* __restrict__ bo, float* __restrict__ out) {
  const int lane = threadIdx.x;
  const int row0 = blockIdx.x << 4;
  const int o0   = blockIdx.y << 6;
  v8f acc[4] = {};
  for (int kk = 0; kk < ED; kk += 32) {
    v16h a = loadA_f16(oh, row0, ED, kk, lane);
#pragma unroll
    for (int nt = 0; nt < 4; ++nt) {
      v16h bt = loadBT_f32(Wo, o0 + nt * 16, ED, kk, lane);
      acc[nt] = wmma16(a, bt, acc[nt]);
    }
  }
#pragma unroll
  for (int nt = 0; nt < 4; ++nt) {
    const int col  = o0 + nt * 16 + (lane & 15);
    const float bb = bo[col];
    const int r0   = row0 + ((lane >> 4) << 3);
#pragma unroll
    for (int j = 0; j < 8; ++j)
      out[(size_t)(r0 + j) * ED + col] = acc[nt][j] + bb;
  }
}

// ---------------- host launcher ------------------------------------------------------
extern "C" void kernel_launch(void* const* d_in, const int* in_sizes, int n_in,
                              void* d_out, int out_size, void* d_ws, size_t ws_size,
                              hipStream_t stream) {
  const float* values = (const float*)d_in[0];
  const float* keys   = (const float*)d_in[1];
  const float* query  = (const float*)d_in[2];
  const int*   mask   = (const int*)d_in[3];
  const float* Wv     = (const float*)d_in[4];
  const float* Wk     = (const float*)d_in[5];
  const float* Wq     = (const float*)d_in[6];
  const float* Wo     = (const float*)d_in[7];
  const float* bo     = (const float*)d_in[8];

  float* out = (float*)d_out;
  float* att = out + (size_t)NB * SEQ * ED;   // attention part of tuple output

  const size_t perBuf = (size_t)NB * NH * SEQ * HD;  // 4,194,304 f16 elems
  _Float16* qh = (_Float16*)d_ws;
  _Float16* kh = qh + perBuf;
  _Float16* vh = kh + perBuf;
  _Float16* oh = qh;  // q-projection is dead after scores; reuse its region

  dim3 w(32);
  dim3 gProj((NB * SEQ) / 16, NH);
  proj_kernel<<<gProj, w, 0, stream>>>(query,  Wq, qh);
  proj_kernel<<<gProj, w, 0, stream>>>(keys,   Wk, kh);
  proj_kernel<<<gProj, w, 0, stream>>>(values, Wv, vh);

  scores_softmax_kernel<<<dim3(SEQ / 16, NB * NH), dim3(256),
                          (size_t)16 * SEQ * sizeof(float), stream>>>(qh, kh, mask, att);
  av_kernel<<<dim3(SEQ / 64, NB * NH), dim3(128), 0, stream>>>(att, vh, oh);
  outproj_kernel<<<dim3((NB * SEQ) / 16, ED / 64), w, 0, stream>>>(oh, Wo, bo, out);
}
